// MultiHeadAttention_52441550684471
// MI455X (gfx1250) — matrix operand order repaired and checked by reading, not compiled, not run on hardware
//
#include <hip/hip_runtime.h>

#define EMBED   1024
#define HEADS   16
#define BATCH   2
#define TLEN    2048
#define DH      64
#define TOTM    (BATCH*TLEN)     // 4096 rows
#define N3C     (3*EMBED)        // 3072

typedef __attribute__((ext_vector_type(16))) __bf16 v16bf;
typedef __attribute__((ext_vector_type(8)))  __bf16 v8bf;
typedef __attribute__((ext_vector_type(8)))  float  v8f;

__device__ __forceinline__ v16bf combine8(v8bf a, v8bf b) {
    v16bf r;
#pragma unroll
    for (int i = 0; i < 8; ++i) { r[i] = a[i]; r[i + 8] = b[i]; }
    return r;
}

__device__ __forceinline__ v8f wmma_bf16(v16bf a, v16bf b, v8f c) {
    // 8 args: (neg_a, A, neg_b, B, c_mod, C, reuse_a, reuse_b)
    return __builtin_amdgcn_wmma_f32_16x16x32_bf16(false, a, false, b,
                                                   (short)0, c, false, false);
}

// ---------------- prep: f32 -> bf16 cast ----------------
__global__ void cvt_f32_bf16(const float* __restrict__ in,
                             __bf16* __restrict__ out, int n) {
    int i = blockIdx.x * blockDim.x + threadIdx.x;
    if (i < n) out[i] = (__bf16)in[i];
}

// ------- prep: coalesced transpose + cast via LDS, W[K,N] -> Wt[N,K] -------
// 256 threads, 32x32 tile; grid = (N/32, K/32). K,N are multiples of 32.
__global__ __launch_bounds__(256) void transpose_cvt(
    const float* __restrict__ W, __bf16* __restrict__ Wt, int K, int N) {
    __shared__ __bf16 tile[32][33];    // +1 pad kills bank conflicts
    int nt = blockIdx.x * 32, kt = blockIdx.y * 32;
    int tx = threadIdx.x & 31, ty = threadIdx.x >> 5;   // ty = 0..7
#pragma unroll
    for (int i = 0; i < 32; i += 8)
        tile[ty + i][tx] = (__bf16)W[(size_t)(kt + ty + i) * N + nt + tx];
    __syncthreads();
#pragma unroll
    for (int i = 0; i < 32; i += 8)
        Wt[(size_t)(nt + ty + i) * K + kt + tx] = tile[tx][ty + i];
}

// ------- QKV GEMM (2x2 register-blocked 32x32 wave tile) -------------------
__global__ __launch_bounds__(256) void qkv_gemm(
    const __bf16* __restrict__ xb,      // [TOTM, EMBED]
    const __bf16* __restrict__ Wt,      // [N3C, EMBED]  (W_kqv^T)
    const float*  __restrict__ bias,    // [N3C]
    __bf16* __restrict__ Qb,            // [B,H,T,DH]
    __bf16* __restrict__ Kb,            // [B,H,T,DH]
    __bf16* __restrict__ Vt)            // [B,H,DH,T]
{
    int lane = threadIdx.x & 31;
    int wave = blockIdx.x * 8 + (threadIdx.x >> 5);
    const int tilesN = N3C / 32;                 // 96
    int m0 = (wave / tilesN) * 32;
    int n0 = (wave % tilesN) * 32;
    int half = lane >> 4, ln = lane & 15;

    const __bf16* ar0 = xb + (size_t)(m0 + ln) * EMBED + half * 16;
    const __bf16* ar1 = ar0 + (size_t)16 * EMBED;
    const __bf16* br0 = Wt + (size_t)(n0 + ln) * EMBED + half * 16;
    const __bf16* br1 = br0 + (size_t)16 * EMBED;

    v8f acc[2][2] = {};
#pragma unroll 2
    for (int k = 0; k < EMBED; k += 32) {
        v16bf a0 = combine8(*(const v8bf*)(ar0 + k), *(const v8bf*)(ar0 + k + 8));
        v16bf a1 = combine8(*(const v8bf*)(ar1 + k), *(const v8bf*)(ar1 + k + 8));
        v16bf b0 = *(const v16bf*)(br0 + k);
        v16bf b1 = *(const v16bf*)(br1 + k);
        acc[0][0] = wmma_bf16(a0, b0, acc[0][0]);
        acc[0][1] = wmma_bf16(a0, b1, acc[0][1]);
        acc[1][0] = wmma_bf16(a1, b0, acc[1][0]);
        acc[1][1] = wmma_bf16(a1, b1, acc[1][1]);
    }

#pragma unroll
    for (int ni = 0; ni < 2; ++ni) {
        int n  = n0 + ni * 16 + ln;
        float bv = bias[n];
        int sec = n >> 10;          // 0:q 1:k 2:v (uniform per wave/ni)
        int c   = n & (EMBED - 1);
        int h   = c >> 6;
        int dh  = c & (DH - 1);
#pragma unroll
        for (int mi = 0; mi < 2; ++mi) {
#pragma unroll
            for (int j = 0; j < 8; ++j) {
                int m = m0 + mi * 16 + j + half * 8;
                int b_ = m >> 11;           // / TLEN
                int t  = m & (TLEN - 1);
                __bf16 v = (__bf16)(acc[mi][ni][j] + bv);
                size_t bh = (size_t)(b_ * HEADS + h);
                if (sec == 0)      Qb[(bh * TLEN + t) * DH + dh] = v;
                else if (sec == 1) Kb[(bh * TLEN + t) * DH + dh] = v;
                else               Vt[(bh * DH + dh) * TLEN + t] = v;
            }
        }
    }
}

// ---------------- flash attention: one wave = 16 query rows ----------------
__global__ __launch_bounds__(256) void attn_kernel(
    const __bf16* __restrict__ Qb, const __bf16* __restrict__ Kb,
    const __bf16* __restrict__ Vt, __bf16* __restrict__ Yb)   // Yb [TOTM, EMBED]
{
    __shared__ __bf16 plds[8 * 16 * 32];   // 1KB per wave P bounce buffer
    int lane = threadIdx.x & 31;
    int wl   = threadIdx.x >> 5;
    int wave = blockIdx.x * 8 + wl;
    int qt = wave & (TLEN / 16 - 1);       // 0..127
    int bh = wave >> 7;                    // 0..31
    int qbase = qt * 16;
    int half = lane >> 4, ln = lane & 15;

    // Preload Q fragments (row-major A frags): dh 0..31 and 32..63
    const __bf16* Qrow = Qb + ((size_t)bh * TLEN + qbase + ln) * DH + half * 16;
    v16bf aq0 = combine8(*(const v8bf*)(Qrow),      *(const v8bf*)(Qrow + 8));
    v16bf aq1 = combine8(*(const v8bf*)(Qrow + 32), *(const v8bf*)(Qrow + 40));

    float mstate[8], lstate[8];
#pragma unroll
    for (int j = 0; j < 8; ++j) { mstate[j] = -1e30f; lstate[j] = 0.f; }
    v8f o0 = {}, o1 = {}, o2 = {}, o3 = {};

    const __bf16* Kbase = Kb + (size_t)bh * TLEN * DH;
    const __bf16* Vbase = Vt + (size_t)bh * DH * TLEN;
    __bf16* pl = plds + wl * 512;

    const int kend = qbase + 16;
    for (int kb = 0; kb < kend; kb += 32) {
        if (kb + 32 < kend)
            __builtin_prefetch(Kbase + (size_t)(kb + 32 + ln) * DH, 0, 0);

        // ---- S = Q K^T  (two 16-key N tiles, K-dim = 64 via 2 wmma each)
        const __bf16* kr0 = Kbase + (size_t)(kb + ln) * DH + half * 16;
        const __bf16* kr1 = Kbase + (size_t)(kb + 16 + ln) * DH + half * 16;
        v8f s0 = {}, s1 = {};
        s0 = wmma_bf16(aq0, *(const v16bf*)(kr0),      s0);
        s0 = wmma_bf16(aq1, *(const v16bf*)(kr0 + 32), s0);
        s1 = wmma_bf16(aq0, *(const v16bf*)(kr1),      s1);
        s1 = wmma_bf16(aq1, *(const v16bf*)(kr1 + 32), s1);

        // ---- online softmax (row lives in one 16-lane half)
        const float scale = 0.125f;   // 1/sqrt(64)
#pragma unroll
        for (int j = 0; j < 8; ++j) {
            int row  = qbase + j + half * 8;
            float v0 = (kb + ln      <= row) ? s0[j] * scale : -1e30f;
            float v1 = (kb + 16 + ln <= row) ? s1[j] * scale : -1e30f;
            float mx = fmaxf(v0, v1);
            mx = fmaxf(mx, __shfl_xor(mx, 1, 32));
            mx = fmaxf(mx, __shfl_xor(mx, 2, 32));
            mx = fmaxf(mx, __shfl_xor(mx, 4, 32));
            mx = fmaxf(mx, __shfl_xor(mx, 8, 32));
            float mnew = fmaxf(mstate[j], mx);
            float p0 = __expf(v0 - mnew);
            float p1 = __expf(v1 - mnew);
            float rs = p0 + p1;
            rs += __shfl_xor(rs, 1, 32);
            rs += __shfl_xor(rs, 2, 32);
            rs += __shfl_xor(rs, 4, 32);
            rs += __shfl_xor(rs, 8, 32);
            float corr = __expf(mstate[j] - mnew);
            lstate[j] = lstate[j] * corr + rs;
            mstate[j] = mnew;
            o0[j] *= corr; o1[j] *= corr; o2[j] *= corr; o3[j] *= corr;
            s0[j] = p0; s1[j] = p1;
        }

        // ---- re-layout P: C/D layout -> A-fragment layout via wave-private LDS
#pragma unroll
        for (int j = 0; j < 8; ++j) {
            int row = j + half * 8;
            pl[row * 32 + ln]      = (__bf16)s0[j];
            pl[row * 32 + ln + 16] = (__bf16)s1[j];
        }
        asm volatile("" ::: "memory");   // same-wave DS ops are in-order
        const __bf16* pr = pl + ln * 32 + half * 16;
        v16bf pf = combine8(*(const v8bf*)(pr), *(const v8bf*)(pr + 8));
        asm volatile("" ::: "memory");

        // ---- O += P @ V  (V^T rows give contiguous B-fragments), K-dim = 32
        const __bf16* vr = Vbase + (size_t)ln * TLEN + kb + half * 16;
        o0 = wmma_bf16(pf, *(const v16bf*)(vr),             o0);
        o1 = wmma_bf16(pf, *(const v16bf*)(vr + 16 * TLEN), o1);
        o2 = wmma_bf16(pf, *(const v16bf*)(vr + 32 * TLEN), o2);
        o3 = wmma_bf16(pf, *(const v16bf*)(vr + 48 * TLEN), o3);
    }

    // ---- normalize and write Y at [b, t, h*64 + dh]
    int b_ = bh >> 4, h = bh & 15;
    size_t outbase = ((size_t)b_ * TLEN) * EMBED + (size_t)h * DH;
#pragma unroll
    for (int j = 0; j < 8; ++j) {
        float inv = 1.0f / lstate[j];
        int t = qbase + j + half * 8;
        size_t ro = outbase + (size_t)t * EMBED;
        Yb[ro + 0  + ln] = (__bf16)(o0[j] * inv);
        Yb[ro + 16 + ln] = (__bf16)(o1[j] * inv);
        Yb[ro + 32 + ln] = (__bf16)(o2[j] * inv);
        Yb[ro + 48 + ln] = (__bf16)(o3[j] * inv);
    }
}

// ------- output projection (2x2 register-blocked): out = Yb @ Wout + b -----
__global__ __launch_bounds__(256) void out_gemm(
    const __bf16* __restrict__ Yb,      // [TOTM, EMBED]
    const __bf16* __restrict__ Wt,      // [EMBED, EMBED] (W_out^T)
    const float*  __restrict__ bias,    // [EMBED]
    float* __restrict__ out)            // [TOTM, EMBED]
{
    int lane = threadIdx.x & 31;
    int wave = blockIdx.x * 8 + (threadIdx.x >> 5);
    const int tilesN = EMBED / 32;               // 32
    int m0 = (wave / tilesN) * 32;
    int n0 = (wave % tilesN) * 32;
    int half = lane >> 4, ln = lane & 15;

    const __bf16* ar0 = Yb + (size_t)(m0 + ln) * EMBED + half * 16;
    const __bf16* ar1 = ar0 + (size_t)16 * EMBED;
    const __bf16* br0 = Wt + (size_t)(n0 + ln) * EMBED + half * 16;
    const __bf16* br1 = br0 + (size_t)16 * EMBED;

    v8f acc[2][2] = {};
#pragma unroll 2
    for (int k = 0; k < EMBED; k += 32) {
        v16bf a0 = combine8(*(const v8bf*)(ar0 + k), *(const v8bf*)(ar0 + k + 8));
        v16bf a1 = combine8(*(const v8bf*)(ar1 + k), *(const v8bf*)(ar1 + k + 8));
        v16bf b0 = *(const v16bf*)(br0 + k);
        v16bf b1 = *(const v16bf*)(br1 + k);
        acc[0][0] = wmma_bf16(a0, b0, acc[0][0]);
        acc[0][1] = wmma_bf16(a0, b1, acc[0][1]);
        acc[1][0] = wmma_bf16(a1, b0, acc[1][0]);
        acc[1][1] = wmma_bf16(a1, b1, acc[1][1]);
    }
#pragma unroll
    for (int ni = 0; ni < 2; ++ni) {
        float bv = bias[n0 + ni * 16 + ln];
#pragma unroll
        for (int mi = 0; mi < 2; ++mi) {
#pragma unroll
            for (int j = 0; j < 8; ++j) {
                int m = m0 + mi * 16 + j + half * 8;
                out[(size_t)m * EMBED + n0 + ni * 16 + ln] = acc[mi][ni][j] + bv;
            }
        }
    }
}

extern "C" void kernel_launch(void* const* d_in, const int* in_sizes, int n_in,
                              void* d_out, int out_size, void* d_ws, size_t ws_size,
                              hipStream_t stream) {
    (void)in_sizes; (void)n_in; (void)out_size; (void)ws_size;
    const float* x     = (const float*)d_in[0];
    const float* Wkqv  = (const float*)d_in[1];
    const float* bkqv  = (const float*)d_in[2];
    const float* Wout  = (const float*)d_in[3];
    const float* bout  = (const float*)d_in[4];

    char* ws = (char*)d_ws;
    size_t off = 0;
    auto carve = [&](size_t bytes) {
        void* p = ws + off;
        off += (bytes + 255) & ~(size_t)255;
        return p;
    };
    __bf16* xb    = (__bf16*)carve((size_t)TOTM * EMBED * 2);   // 8MB
    __bf16* WkqvT = (__bf16*)carve((size_t)N3C  * EMBED * 2);   // 6MB
    __bf16* WoutT = (__bf16*)carve((size_t)EMBED* EMBED * 2);   // 2MB
    __bf16* Qb    = (__bf16*)carve((size_t)TOTM * EMBED * 2);   // 8MB
    __bf16* Kb    = (__bf16*)carve((size_t)TOTM * EMBED * 2);   // 8MB
    __bf16* Vt    = (__bf16*)carve((size_t)TOTM * EMBED * 2);   // 8MB
    __bf16* Yb    = (__bf16*)carve((size_t)TOTM * EMBED * 2);   // 8MB

    int nx = TOTM * EMBED;                       // 4,194,304
    cvt_f32_bf16<<<(nx + 255) / 256, 256, 0, stream>>>(x, xb, nx);
    transpose_cvt<<<dim3(N3C / 32, EMBED / 32), 256, 0, stream>>>(Wkqv, WkqvT, EMBED, N3C);
    transpose_cvt<<<dim3(EMBED / 32, EMBED / 32), 256, 0, stream>>>(Wout, WoutT, EMBED, EMBED);

    // (TOTM/32)*(N3C/32) = 12288 wave-tiles, 8 waves per 256-thread block
    qkv_gemm<<<12288 / 8, 256, 0, stream>>>(xb, WkqvT, bkqv, Qb, Kb, Vt);

    // B*H*(T/16) = 4096 waves -> 512 blocks
    attn_kernel<<<4096 / 8, 256, 0, stream>>>(Qb, Kb, Vt, Yb);

    // (TOTM/32)*(EMBED/32) = 4096 wave-tiles -> 512 blocks
    out_gemm<<<4096 / 8, 256, 0, stream>>>(Yb, WoutT, bout, (float*)d_out);
}
